// NERRelationModel_53249004536066
// MI455X (gfx1250) — matrix operand order repaired and checked by reading, not compiled, not run on hardware
//
#include <hip/hip_runtime.h>
#include <hip/hip_bf16.h>

typedef _Float16 half_t;
typedef __attribute__((ext_vector_type(8)))  _Float16 v8h;
typedef __attribute__((ext_vector_type(16))) _Float16 v16h;
typedef __attribute__((ext_vector_type(8)))  float    v8f;

// D = A(16x32 f16) * B(32x16 f16) + C(16x16 f32)
#define WMMA_F16(a, b, c) \
  __builtin_amdgcn_wmma_f32_16x16x32_f16(false, (a), false, (b), (short)0, (c), false, false)

// ---------------------------------------------------------------------------
// Fragment loaders (wave32), contiguous k = 16*(lane/16) + s for A and B, vector-load friendly.
// A row-major (lda = K): lane m = lane&15; lanes 0-15 K=0..15,
// lanes 16-31 K=16..31.  Two aligned 16B loads + shuffle.
__device__ __forceinline__ v16h load_a_frag(const half_t* Arow_k, int k0) {
  // Arow_k already points at A + m*lda + khi_a; k0 multiple of 32
  v8h lo = *(const v8h*)(Arow_k + k0);        // halves khi..khi+7
  v8h hi = *(const v8h*)(Arow_k + k0 + 8);    // halves khi+8..khi+15
  return __builtin_shufflevector(lo, hi, 0, 1, 2, 3, 4, 5, 6, 7,
                                 8, 9, 10, 11, 12, 13, 14, 15);
}

// B stored TRANSPOSED (ldb = K, row n holds column n of logical B):
// lane n = lane&15; lanes 0-15 K=0..15, lanes 16-31 K=16..31 -> one 32B load.
__device__ __forceinline__ v16h load_bt_frag(const half_t* Brow_k, int k0) {
  return *(const v16h*)(Brow_k + k0);
}

// ---------------------------------------------------------------------------
// Stage 1: attention pooling over W=8 tokens + entity type embedding
__global__ __launch_bounds__(128) void pool_kernel(
    const float* __restrict__ seq, const int* __restrict__ starts,
    const int* __restrict__ type_ids, const float* __restrict__ type_emb,
    float* __restrict__ x0) {
  __shared__ float toks[8 * 768];
  __shared__ float red[128];
  __shared__ float logits[8];
  int be = blockIdx.x;
  int b = be >> 7;  // E = 128
  int t = threadIdx.x;
  int start = starts[be];
  const float* base = seq + ((size_t)b * 1024 + start) * 768;  // W rows contiguous
  for (int i = t; i < 8 * 768; i += 128) toks[i] = base[i];
  __syncthreads();
  float mean[6];
#pragma unroll
  for (int j = 0; j < 6; ++j) {
    int c = t + 128 * j;
    float s = 0.f;
#pragma unroll
    for (int w = 0; w < 8; ++w) s += toks[w * 768 + c];
    mean[j] = s * 0.125f;
  }
  for (int w = 0; w < 8; ++w) {
    float p = 0.f;
#pragma unroll
    for (int j = 0; j < 6; ++j) { int c = t + 128 * j; p += toks[w * 768 + c] * mean[j]; }
    red[t] = p; __syncthreads();
    for (int s = 64; s > 0; s >>= 1) { if (t < s) red[t] += red[t + s]; __syncthreads(); }
    if (t == 0) logits[w] = red[0];
    __syncthreads();
  }
  float mx = logits[0];
#pragma unroll
  for (int w = 1; w < 8; ++w) mx = fmaxf(mx, logits[w]);
  float wt[8]; float sum = 0.f;
#pragma unroll
  for (int w = 0; w < 8; ++w) { wt[w] = expf(logits[w] - mx); sum += wt[w]; }
  float inv = 1.f / sum;
  int ty = type_ids[be];
  float* orow = x0 + (size_t)be * 768;
#pragma unroll
  for (int j = 0; j < 6; ++j) {
    int c = t + 128 * j;
    float s = 0.f;
#pragma unroll
    for (int w = 0; w < 8; ++w) s += wt[w] * inv * toks[w * 768 + c];
    orow[c] = s + type_emb[ty * 768 + c];
  }
}

// ---------------------------------------------------------------------------
__global__ void cast_f16(const float* __restrict__ in, half_t* __restrict__ out, int n) {
  int i = blockIdx.x * blockDim.x + threadIdx.x;
  if (i < n) out[i] = (half_t)in[i];
}

// transpose+cast weight: in (R,C) f32 row-major -> out (C,R) f16
__global__ void transpose_cast_w(const float* __restrict__ in, half_t* __restrict__ out,
                                 int R, int C) {
  int idx = blockIdx.x * blockDim.x + threadIdx.x;
  if (idx >= R * C) return;
  int c = idx / R, r = idx - c * R;
  out[idx] = (half_t)in[r * C + c];   // out[c*R + r]
}

// batched transpose+cast for GAT features: in (B*128, HD) -> out (B, HD, 128)
__global__ void transpose_cast_h(const float* __restrict__ in, half_t* __restrict__ out,
                                 int HD) {
  int idx = blockIdx.x * blockDim.x + threadIdx.x;
  int per_b = HD * 128;
  int b = idx / per_b;
  if (b >= 32) return;
  int rem = idx - b * per_b;
  int c = rem >> 7;          // 0..HD-1
  int e = rem & 127;
  out[idx] = (half_t)in[(size_t)(b * 128 + e) * HD + c];
}

// ---------------------------------------------------------------------------
// f16 WMMA GEMM: C(MxN,f32) = A(MxK,row-major) * Bt(NxK,transposed).
// 128 threads = 4 waves; one 16x64 strip (4 wmma tiles) per wave.
// grid*4 == (M/16)*(N/64); no remainders.
__global__ __launch_bounds__(128) void gemm_f16_wmma(
    const half_t* __restrict__ A, const half_t* __restrict__ Bt,
    float* __restrict__ C, int M, int N, int K) {
  int lane = threadIdx.x & 31;
  int wave = threadIdx.x >> 5;
  int ngrp = N >> 6;
  int job = blockIdx.x * 4 + wave;
  int tm = (job / ngrp) << 4;
  int tn = (job % ngrp) << 6;
  int n0 = lane & 15;
  int khi_a = (lane >> 4) * 16;
  int khi_b = (lane >> 4) * 16;
  const half_t* Arow = A + (size_t)(tm + n0) * K + khi_a;
  const half_t* B0 = Bt + (size_t)(tn + n0) * K + khi_b;
  const half_t* B1 = B0 + (size_t)16 * K;
  const half_t* B2 = B0 + (size_t)32 * K;
  const half_t* B3 = B0 + (size_t)48 * K;
  v8f acc0 = {}, acc1 = {}, acc2 = {}, acc3 = {};
  for (int k0 = 0; k0 < K; k0 += 32) {
    __builtin_prefetch(Arow + k0 + 256, 0, 1);
    __builtin_prefetch(B0 + k0 + 256, 0, 1);
    v16h a  = load_a_frag(Arow, k0);
    v16h b0 = load_bt_frag(B0, k0);
    v16h b1 = load_bt_frag(B1, k0);
    v16h b2 = load_bt_frag(B2, k0);
    v16h b3 = load_bt_frag(B3, k0);
    acc0 = WMMA_F16(a, b0, acc0);
    acc1 = WMMA_F16(a, b1, acc1);
    acc2 = WMMA_F16(a, b2, acc2);
    acc3 = WMMA_F16(a, b3, acc3);
  }
  int mo = (lane >> 4) * 8;
#pragma unroll
  for (int r = 0; r < 8; ++r) {
    float* cr = C + (size_t)(tm + mo + r) * N + n0;
    cr[tn]      = acc0[r];
    cr[tn + 16] = acc1[r];
    cr[tn + 32] = acc2[r];
    cr[tn + 48] = acc3[r];
  }
}

// ---------------------------------------------------------------------------
// GAT attention + aggregation: one block per (batch, head).
// alpha (128x128 f16) built in LDS; WMMA alpha @ h with h transposed (B,HD,128).
__global__ __launch_bounds__(128) void gat_attn_kernel(
    const float* __restrict__ hbuf, const half_t* __restrict__ ht,
    const float* __restrict__ a_src, const float* __restrict__ a_dst,
    float* __restrict__ obuf, int D, int HD, int heads) {
  int b  = blockIdx.x / heads;
  int hh = blockIdx.x % heads;
  int col0 = hh * D;
  __shared__ float  s_s[128];
  __shared__ float  s_d[128];
  __shared__ half_t alpha[128 * 128];  // 32 KB
  int t = threadIdx.x;
  {
    const float* rp = hbuf + (size_t)(b * 128 + t) * HD + col0;
    float ss = 0.f, dd = 0.f;
    for (int k = 0; k < D; ++k) {
      float v = rp[k];
      ss += v * a_src[hh * D + k];
      dd += v * a_dst[hh * D + k];
    }
    s_s[t] = ss; s_d[t] = dd;
  }
  __syncthreads();
  {
    float di = s_d[t];
    float mx = -1e30f;
    for (int j = 0; j < 128; ++j) {
      float e = di + s_s[j]; e = (e > 0.f) ? e : 0.2f * e;
      mx = fmaxf(mx, e);
    }
    float sum = 0.f;
    for (int j = 0; j < 128; ++j) {
      float e = di + s_s[j]; e = (e > 0.f) ? e : 0.2f * e;
      float ex = expf(e - mx);
      sum += ex;
      alpha[t * 128 + j] = (half_t)ex;
    }
    float inv = 1.f / sum;
    for (int j = 0; j < 128; ++j)
      alpha[t * 128 + j] = (half_t)((float)alpha[t * 128 + j] * inv);
  }
  __syncthreads();
  int lane = t & 31, wave = t >> 5;
  int n0 = lane & 15;
  int khi_a = (lane >> 4) * 16;
  int khi_b = (lane >> 4) * 16;
  int ngrp = D >> 6;                    // 2 (layer1) or 1 (layer2)
  int njobs = 8 * ngrp;                 // 16 or 8: multiple of 4
  const half_t* Bbase = ht + ((size_t)b * HD + col0) * 128;
  for (int job = wave; job < njobs; job += 4) {
    int tm = (job / ngrp) << 4;
    int tn = (job % ngrp) << 6;
    const half_t* Arow = alpha + (tm + n0) * 128 + khi_a;
    const half_t* B0 = Bbase + (size_t)(tn + n0) * 128 + khi_b;
    const half_t* B1 = B0 + 16 * 128;
    const half_t* B2 = B0 + 32 * 128;
    const half_t* B3 = B0 + 48 * 128;
    v8f acc0 = {}, acc1 = {}, acc2 = {}, acc3 = {};
    for (int k0 = 0; k0 < 128; k0 += 32) {
      v16h a  = load_a_frag(Arow, k0);
      v16h b0 = load_bt_frag(B0, k0);
      v16h b1 = load_bt_frag(B1, k0);
      v16h b2 = load_bt_frag(B2, k0);
      v16h b3 = load_bt_frag(B3, k0);
      acc0 = WMMA_F16(a, b0, acc0);
      acc1 = WMMA_F16(a, b1, acc1);
      acc2 = WMMA_F16(a, b2, acc2);
      acc3 = WMMA_F16(a, b3, acc3);
    }
    int mo = (lane >> 4) * 8;
#pragma unroll
    for (int r = 0; r < 8; ++r) {
      float* orow = obuf + (size_t)(b * 128 + tm + mo + r) * HD + col0 + n0;
      orow[tn]      = acc0[r];
      orow[tn + 16] = acc1[r];
      orow[tn + 32] = acc2[r];
      orow[tn + 48] = acc3[r];
    }
  }
}

// ---------------------------------------------------------------------------
// Fused (x + bias) -> LayerNorm -> ELU, optional f32 and f16 outputs.
__global__ void ln_elu_kernel(const float* __restrict__ in, const float* __restrict__ bias,
                              const float* __restrict__ gamma, const float* __restrict__ beta,
                              float* __restrict__ outf, half_t* __restrict__ outh, int N) {
  __shared__ float red[256];
  __shared__ float stats[2];
  int row = blockIdx.x, t = threadIdx.x;
  float v = in[(size_t)row * N + t] + bias[t];
  red[t] = v; __syncthreads();
  for (int s = N >> 1; s > 0; s >>= 1) { if (t < s) red[t] += red[t + s]; __syncthreads(); }
  if (t == 0) stats[0] = red[0] / N;
  __syncthreads();
  float m = stats[0];
  float d = v - m;
  red[t] = d * d; __syncthreads();
  for (int s = N >> 1; s > 0; s >>= 1) { if (t < s) red[t] += red[t + s]; __syncthreads(); }
  if (t == 0) stats[1] = red[0] / N;
  __syncthreads();
  float y = d * rsqrtf(stats[1] + 1e-5f) * gamma[t] + beta[t];
  float e = (y > 0.f) ? y : (expf(y) - 1.f);
  if (outf) outf[(size_t)row * N + t] = e;
  if (outh) outh[(size_t)row * N + t] = (half_t)e;
}

// ---------------------------------------------------------------------------
// Build pair vectors pv = [x1(64) | x2(64) | rel(32) | cls(768)] as f16.
__global__ void build_pv(const float* __restrict__ x2, const int* __restrict__ pair_idx,
                         const int* __restrict__ rel_ids, const float* __restrict__ rel_emb,
                         const float* __restrict__ seq, half_t* __restrict__ pvh, int total) {
  int idx = blockIdx.x * blockDim.x + threadIdx.x;
  if (idx >= total) return;
  int m = idx / 928, k = idx - m * 928;
  int b = m >> 9;        // P = 512
  int p = m & 511;
  float v;
  if (k < 64)       v = x2[(size_t)(b * 128 + pair_idx[(b * 512 + p) * 2]) * 64 + k];
  else if (k < 128) v = x2[(size_t)(b * 128 + pair_idx[(b * 512 + p) * 2 + 1]) * 64 + (k - 64)];
  else if (k < 160) v = rel_emb[rel_ids[b * 512 + p] * 32 + (k - 128)];
  else              v = seq[(size_t)b * 1024 * 768 + (k - 160)];  // cls = seq[b,0,:]
  pvh[idx] = (half_t)v;
}

__global__ void init_scores(float* __restrict__ out, const float* __restrict__ br2, int n) {
  int i = blockIdx.x * blockDim.x + threadIdx.x;
  if (i < n) out[i] = br2[0];
}

// ---------------------------------------------------------------------------
// Pair MLP: hidden = relu(pv @ Wr1 + br1); scores += hidden @ Wr2
// M=16384, N=256, K=928; 16x64 strip per wave, fused reduce-over-N epilogue.
__global__ __launch_bounds__(128) void pair_mlp(
    const half_t* __restrict__ A, const half_t* __restrict__ Bt,
    const float* __restrict__ br1, const float* __restrict__ Wr2,
    float* __restrict__ scores) {
  int lane = threadIdx.x & 31;
  int wave = threadIdx.x >> 5;
  int job = blockIdx.x * 4 + wave;     // 4096 jobs
  int tm = (job >> 2) << 4;            // ngrp = 256/64 = 4
  int tn = (job & 3) << 6;
  int n0 = lane & 15;
  int khi_a = (lane >> 4) * 16;
  int khi_b = (lane >> 4) * 16;
  const half_t* Arow = A + (size_t)(tm + n0) * 928 + khi_a;
  const half_t* B0 = Bt + (size_t)(tn + n0) * 928 + khi_b;
  const half_t* B1 = B0 + (size_t)16 * 928;
  const half_t* B2 = B0 + (size_t)32 * 928;
  const half_t* B3 = B0 + (size_t)48 * 928;
  v8f acc0 = {}, acc1 = {}, acc2 = {}, acc3 = {};
  for (int k0 = 0; k0 < 928; k0 += 32) {
    __builtin_prefetch(Arow + k0 + 256, 0, 1);
    __builtin_prefetch(B0 + k0 + 256, 0, 1);
    v16h a  = load_a_frag(Arow, k0);
    v16h b0 = load_bt_frag(B0, k0);
    v16h b1 = load_bt_frag(B1, k0);
    v16h b2 = load_bt_frag(B2, k0);
    v16h b3 = load_bt_frag(B3, k0);
    acc0 = WMMA_F16(a, b0, acc0);
    acc1 = WMMA_F16(a, b1, acc1);
    acc2 = WMMA_F16(a, b2, acc2);
    acc3 = WMMA_F16(a, b3, acc3);
  }
  float br0 = br1[tn + n0],      w0 = Wr2[tn + n0];
  float brA = br1[tn + 16 + n0], w1 = Wr2[tn + 16 + n0];
  float brB = br1[tn + 32 + n0], w2 = Wr2[tn + 32 + n0];
  float brC = br1[tn + 48 + n0], w3 = Wr2[tn + 48 + n0];
  int mo = (lane >> 4) * 8;
#pragma unroll
  for (int r = 0; r < 8; ++r) {
    float v = fmaxf(acc0[r] + br0, 0.f) * w0
            + fmaxf(acc1[r] + brA, 0.f) * w1
            + fmaxf(acc2[r] + brB, 0.f) * w2
            + fmaxf(acc3[r] + brC, 0.f) * w3;
    atomicAdd(&scores[tm + mo + r], v);
  }
}

// ---------------------------------------------------------------------------
// Workspace layout (bytes); PVH/WR1T alias dead stage-1/2 buffers.
static const size_t OFF_X0   = 0;                                  // 4096*768*4
static const size_t OFF_X0H  = OFF_X0  + 4096ull * 768 * 4;
static const size_t OFF_W1T  = OFF_X0H + 4096ull * 768 * 2;        // (256,768) f16
static const size_t OFF_H1   = OFF_W1T + 768ull  * 256 * 2;
static const size_t OFF_H1T  = OFF_H1  + 4096ull * 256 * 4;        // (32,256,128) f16
static const size_t OFF_O1   = OFF_H1T + 4096ull * 256 * 2;
static const size_t OFF_X1H  = OFF_O1  + 4096ull * 256 * 4;
static const size_t OFF_H2   = OFF_X1H + 4096ull * 256 * 2;
static const size_t OFF_H2T  = OFF_H2  + 4096ull * 64 * 4;         // (32,64,128) f16
static const size_t OFF_W2T  = OFF_H2T + 4096ull * 64 * 2;         // (64,256) f16
static const size_t OFF_O2   = OFF_W2T + 256ull  * 64 * 2;
static const size_t OFF_X2   = OFF_O2  + 4096ull * 64 * 4;
static const size_t OFF_PVH  = 0;                                  // 16384*928*2 aliases X0..O1
static const size_t OFF_WR1T = OFF_PVH + 16384ull * 928 * 2;       // (256,928) aliases X1H (dead)

extern "C" void kernel_launch(void* const* d_in, const int* in_sizes, int n_in,
                              void* d_out, int out_size, void* d_ws, size_t ws_size,
                              hipStream_t stream) {
  (void)in_sizes; (void)n_in; (void)out_size; (void)ws_size;
  const float* seq      = (const float*)d_in[0];
  const int*   starts   = (const int*)  d_in[1];
  const int*   type_ids = (const int*)  d_in[2];
  const int*   pair_idx = (const int*)  d_in[3];
  const int*   rel_ids  = (const int*)  d_in[4];
  const float* type_emb = (const float*)d_in[5];
  const float* rel_emb  = (const float*)d_in[6];
  const float* W1       = (const float*)d_in[7];
  const float* a_src1   = (const float*)d_in[8];
  const float* a_dst1   = (const float*)d_in[9];
  const float* b1       = (const float*)d_in[10];
  const float* ln1_g    = (const float*)d_in[11];
  const float* ln1_b    = (const float*)d_in[12];
  const float* W2       = (const float*)d_in[13];
  const float* a_src2   = (const float*)d_in[14];
  const float* a_dst2   = (const float*)d_in[15];
  const float* b2       = (const float*)d_in[16];
  const float* ln2_g    = (const float*)d_in[17];
  const float* ln2_b    = (const float*)d_in[18];
  const float* Wr1      = (const float*)d_in[19];
  const float* br1      = (const float*)d_in[20];
  const float* Wr2      = (const float*)d_in[21];
  const float* br2      = (const float*)d_in[22];

  char* ws = (char*)d_ws;
  float*  x0   = (float*) (ws + OFF_X0);
  half_t* x0h  = (half_t*)(ws + OFF_X0H);
  half_t* W1t  = (half_t*)(ws + OFF_W1T);
  float*  h1   = (float*) (ws + OFF_H1);
  half_t* h1t  = (half_t*)(ws + OFF_H1T);
  float*  o1   = (float*) (ws + OFF_O1);
  half_t* x1h  = (half_t*)(ws + OFF_X1H);
  float*  h2   = (float*) (ws + OFF_H2);
  half_t* h2t  = (half_t*)(ws + OFF_H2T);
  half_t* W2t  = (half_t*)(ws + OFF_W2T);
  float*  o2   = (float*) (ws + OFF_O2);
  float*  x2   = (float*) (ws + OFF_X2);
  half_t* pvh  = (half_t*)(ws + OFF_PVH);
  half_t* Wr1t = (half_t*)(ws + OFF_WR1T);
  float*  scores = (float*)d_out;

  // Stage 1: pooling + type embedding
  pool_kernel<<<4096, 128, 0, stream>>>(seq, starts, type_ids, type_emb, x0);

  // Stage 2: GAT layer 1
  cast_f16<<<(3145728 + 255) / 256, 256, 0, stream>>>(x0, x0h, 3145728);
  transpose_cast_w<<<(196608 + 255) / 256, 256, 0, stream>>>(W1, W1t, 768, 256);
  gemm_f16_wmma<<<256, 128, 0, stream>>>(x0h, W1t, h1, 4096, 256, 768);   // 1024 jobs
  transpose_cast_h<<<(1048576 + 255) / 256, 256, 0, stream>>>(h1, h1t, 256);
  gat_attn_kernel<<<64, 128, 0, stream>>>(h1, h1t, a_src1, a_dst1, o1, 128, 256, 2);
  ln_elu_kernel<<<4096, 256, 0, stream>>>(o1, b1, ln1_g, ln1_b, nullptr, x1h, 256);

  // Stage 3: GAT layer 2
  transpose_cast_w<<<(16384 + 255) / 256, 256, 0, stream>>>(W2, W2t, 256, 64);
  gemm_f16_wmma<<<64, 128, 0, stream>>>(x1h, W2t, h2, 4096, 64, 256);     // 256 jobs
  transpose_cast_h<<<(262144 + 255) / 256, 256, 0, stream>>>(h2, h2t, 64);
  gat_attn_kernel<<<32, 128, 0, stream>>>(h2, h2t, a_src2, a_dst2, o2, 64, 64, 1);
  ln_elu_kernel<<<4096, 64, 0, stream>>>(o2, b2, ln2_g, ln2_b, x2, nullptr, 64);

  // Stage 4: pair scoring MLP
  transpose_cast_w<<<(237568 + 255) / 256, 256, 0, stream>>>(Wr1, Wr1t, 928, 256);
  int pv_total = 16384 * 928;
  build_pv<<<(pv_total + 255) / 256, 256, 0, stream>>>(x2, pair_idx, rel_ids, rel_emb, seq,
                                                       pvh, pv_total);
  init_scores<<<64, 256, 0, stream>>>(scores, br2, 16384);
  pair_mlp<<<1024, 128, 0, stream>>>(pvh, Wr1t, br1, Wr2, scores);        // 4096 jobs
}
